// LocalAstrometryMatcher_2233382994710
// MI455X (gfx1250) — matrix operand order repaired and checked by reading, not compiled, not run on hardware
//
#include <hip/hip_runtime.h>
#include <hip/hip_bf16.h>
#include <math.h>

typedef __attribute__((ext_vector_type(16))) _Float16 v16h;
typedef __attribute__((ext_vector_type(8)))  float    v8f;

#define NPIX   4096          // 64*64
#define NELEM  16777216      // 128*64*64*32
#define NBPIX  524288        // 128*4096

__device__ __forceinline__ float gelu_erf(float x) {
    return 0.5f * x * (1.0f + erff(x * 0.70710678118654752f));
}

__device__ __forceinline__ float gauss_raw(int p) {
    int y = p >> 6, x = p & 63;
    float yy = -1.0f + (2.0f / 63.0f) * (float)y;
    float xx = -1.0f + (2.0f / 63.0f) * (float)x;
    return expf(-(yy * yy + xx * xx) * 2.0f);   // /(2*0.5^2) == *2
}

// ---------------------------------------------------------------------------
// Weight packing: f32 OIHW -> f16 WMMA A-matrix lane layout.
// Conv layers l=0..5 (rubin 0..2, vis 3..5): [mt][tap][lane][e] per layer (9216 halves).
// Proj pr=0,1: [mt][lane][e] (1024 halves each) at offset 55296.
// ---------------------------------------------------------------------------
__global__ void pack_kernel(const float* __restrict__ rc, const float* __restrict__ vc,
                            const float* __restrict__ rp, const float* __restrict__ vp,
                            _Float16* __restrict__ pw) {
    int i = blockIdx.x * 256 + threadIdx.x;
    if (i >= 57344) return;
    int e = i & 15;
    int lane = (i >> 4) & 31;
    int hs = lane >> 4;
    int v = e >> 1, h = e & 1;
    int K = 16 * hs + 2 * v + h;   // contiguous k order (k = 16*hs + e), matches B fill
    float val;
    if (i < 55296) {
        int tap = (i >> 9) % 9;
        int mt  = (i / 4608) & 1;
        int l   = i / 9216;
        int cout = mt * 16 + (lane & 15);
        const float* src = (l < 3) ? rc : vc;
        int ll = (l < 3) ? l : l - 3;
        val = src[((ll * 32 + cout) * 32 + K) * 9 + tap];
    } else {
        int r = i - 55296;
        int mt = (r >> 9) & 1;
        int pr = r >> 10;
        int cout = mt * 16 + (lane & 15);
        val = ((pr == 0) ? rp : vp)[cout * 32 + K];
    }
    pw[i] = (_Float16)val;
}

// ---------------------------------------------------------------------------
// conv0: small-cin 3x3 conv, NCHW f32 in -> channels-last f32 out
// ---------------------------------------------------------------------------
__global__ void conv0_kernel(const float* __restrict__ in, const float* __restrict__ w,
                             int cin, float* __restrict__ out) {
    int i = blockIdx.x * 256 + threadIdx.x;       // over 128*64*64*32
    int co = i & 31;
    int x  = (i >> 5) & 63;
    int y  = (i >> 11) & 63;
    int b  = i >> 17;
    float acc = 0.0f;
    for (int ci = 0; ci < cin; ++ci)
        for (int ky = 0; ky < 3; ++ky) {
            int yy = y + ky - 1;
            if (yy < 0 || yy > 63) continue;
            for (int kx = 0; kx < 3; ++kx) {
                int xx = x + kx - 1;
                if (xx < 0 || xx > 63) continue;
                acc += in[(((size_t)b * cin + ci) * 64 + yy) * 64 + xx]
                     * w[((co * cin + ci) * 3 + ky) * 3 + kx];
            }
        }
    out[i] = acc;   // i == ((b*64+y)*64+x)*32+co  (channels-last)
}

// ---------------------------------------------------------------------------
// WMMA implicit-GEMM conv: K=cin=32 per tap, M=32 couts (2 tiles), N=16 pixels.
// NTAPS=9 -> 3x3 conv (zero pad); NTAPS=1 -> 32x32 projection GEMM.
// ---------------------------------------------------------------------------
template <int NTAPS>
__global__ __launch_bounds__(256) void conv_wmma_kernel(
        const _Float16* __restrict__ act,     // [B][64][64][32] f16
        const _Float16* __restrict__ pw,      // packed A for this layer
        float* __restrict__ out)              // [B][64][64][32] f32
{
    __shared__ __align__(32) _Float16 sW[2 * NTAPS * 512];
    const int tid = threadIdx.x;
    {
        const int n16 = (2 * NTAPS * 512) / 8;   // uint4 chunks
        const uint4* s = (const uint4*)pw;
        uint4* d = (uint4*)sW;
        for (int i = tid; i < n16; i += 256) d[i] = s[i];
    }
    __syncthreads();

    const int wid = tid >> 5, lane = tid & 31;
    const int gt = blockIdx.x * 8 + wid;          // tile id over 32768
    const int tx = gt & 3;
    const int y  = (gt >> 2) & 63;
    const int b  = gt >> 8;
    const int n  = lane & 15;
    const int kh = lane >> 4;
    const int x  = tx * 16 + n;
    const int c0 = kh * 16;

    v8f acc0 = {};
    v8f acc1 = {};
#pragma unroll
    for (int tap = 0; tap < NTAPS; ++tap) {
        const int dy = (NTAPS == 9) ? (tap / 3 - 1) : 0;
        const int dx = (NTAPS == 9) ? (tap % 3 - 1) : 0;
        const int yy = y + dy, xx = x + dx;
        v16h bm = {};
        if (yy >= 0 && yy < 64 && xx >= 0 && xx < 64)
            bm = *(const v16h*)(act + ((((size_t)b * 64 + yy) * 64 + xx) << 5) + c0);
        v16h a0 = *(const v16h*)(sW + tap * 512 + lane * 16);
        v16h a1 = *(const v16h*)(sW + (NTAPS + tap) * 512 + lane * 16);
        acc0 = __builtin_amdgcn_wmma_f32_16x16x32_f16(false, a0, false, bm, (short)0, acc0, false, false);
        acc1 = __builtin_amdgcn_wmma_f32_16x16x32_f16(false, a1, false, bm, (short)0, acc1, false, false);
    }

    float* op = out + ((((size_t)b * 64 + y) * 64 + tx * 16 + n) << 5);
#pragma unroll
    for (int r = 0; r < 8; ++r) {          // C/D layout: lanes 0-15 M=r, 16-31 M=r+8
        op[8 * kh + r]      = acc0[r];
        op[16 + 8 * kh + r] = acc1[r];
    }
}

// ---------------------------------------------------------------------------
// GroupNorm stats: per (b, group of 8 ch) mean + rstd over 8*4096 values
// ---------------------------------------------------------------------------
__global__ void gn_stats_kernel(const float* __restrict__ x, float* __restrict__ stats) {
    __shared__ float ssum[256], ssq[256];
    int bg = blockIdx.x;
    int b = bg >> 2, g = bg & 3;
    int c0 = g * 8;
    float s = 0.0f, q = 0.0f;
    for (int t = threadIdx.x; t < 8 * NPIX; t += 256) {
        int p = t >> 3;
        float v = x[((size_t)b * NPIX + p) * 32 + c0 + (t & 7)];
        s += v; q += v * v;
    }
    ssum[threadIdx.x] = s; ssq[threadIdx.x] = q;
    __syncthreads();
    for (int off = 128; off > 0; off >>= 1) {
        if (threadIdx.x < off) {
            ssum[threadIdx.x] += ssum[threadIdx.x + off];
            ssq[threadIdx.x]  += ssq[threadIdx.x + off];
        }
        __syncthreads();
    }
    if (threadIdx.x == 0) {
        float mean = ssum[0] * (1.0f / 32768.0f);
        float var  = ssq[0] * (1.0f / 32768.0f) - mean * mean;
        stats[bg * 2]     = mean;
        stats[bg * 2 + 1] = rsqrtf(var + 1e-5f);
    }
}

__global__ void gn_apply_kernel(const float* __restrict__ x, const float* __restrict__ stats,
                                const float* __restrict__ gamma, const float* __restrict__ beta,
                                _Float16* __restrict__ dst) {
    int i = blockIdx.x * 256 + threadIdx.x;    // over NELEM
    int c = i & 31;
    int b = i >> 17;
    int g = c >> 3;
    float mean = stats[(b * 4 + g) * 2];
    float rstd = stats[(b * 4 + g) * 2 + 1];
    float v = (x[i] - mean) * rstd * gamma[c] + beta[c];
    dst[i] = (_Float16)gelu_erf(v);
}

// ---------------------------------------------------------------------------
// per-pixel l2-norm over 32 channels; optional energy output
// ---------------------------------------------------------------------------
__global__ void l2norm_kernel(const float* __restrict__ x, _Float16* __restrict__ dst,
                              float* __restrict__ energy) {
    int p = blockIdx.x * 256 + threadIdx.x;    // over NBPIX
    const float* px = x + (size_t)p * 32;
    float s = 0.0f;
#pragma unroll
    for (int c = 0; c < 32; ++c) s += px[c] * px[c];
    float nrm = sqrtf(s);
    float inv = 1.0f / fmaxf(nrm, 1e-6f);
#pragma unroll
    for (int c = 0; c < 32; ++c) dst[(size_t)p * 32 + c] = (_Float16)(px[c] * inv);
    if (energy) energy[p] = s * inv * inv;
}

__global__ void gsum_kernel(float* __restrict__ out) {
    __shared__ float sm[256];
    float s = 0.0f;
    for (int p = threadIdx.x; p < NPIX; p += 256) s += gauss_raw(p);
    sm[threadIdx.x] = s; __syncthreads();
    for (int off = 128; off > 0; off >>= 1) {
        if (threadIdx.x < off) sm[threadIdx.x] += sm[threadIdx.x + off];
        __syncthreads();
    }
    if (threadIdx.x == 0) out[0] = sm[0];
}

__global__ void swsum_kernel(const float* __restrict__ energy, const float* __restrict__ gsum,
                             float* __restrict__ swsum) {
    __shared__ float sm[256];
    int b = blockIdx.x;
    float gs = gsum[0];
    float s = 0.0f;
    for (int p = threadIdx.x; p < NPIX; p += 256)
        s += energy[b * NPIX + p] * (gauss_raw(p) / gs);
    sm[threadIdx.x] = s; __syncthreads();
    for (int off = 128; off > 0; off >>= 1) {
        if (threadIdx.x < off) sm[threadIdx.x] += sm[threadIdx.x + off];
        __syncthreads();
    }
    if (threadIdx.x == 0) swsum[b] = sm[0];
}

__global__ void sw_kernel(const float* __restrict__ energy, const float* __restrict__ gsum,
                          const float* __restrict__ swsum, float* __restrict__ w2) {
    int i = blockIdx.x * 256 + threadIdx.x;    // over NBPIX
    int b = i >> 12, p = i & 4095;
    float sw = energy[i] * (gauss_raw(p) / gsum[0]) / (swsum[b] + 1e-8f);
    w2[i] = sw + 1e-10f;                       // (sqrt(sw+1e-10))^2 applied twice in cost
}

// ---------------------------------------------------------------------------
// cost volume: logits[b, s] = scale * sum_p w2[b,p] * <rubinN[b,p,:], visN[b,clamp(p+d),:]>
// ---------------------------------------------------------------------------
__global__ void cost_kernel(const _Float16* __restrict__ rn, const _Float16* __restrict__ vn,
                            const float* __restrict__ w2, float* __restrict__ logits) {
    __shared__ float sm[256];
    int bs = blockIdx.x;
    int b = bs / 49, s = bs % 49;
    int dy = s / 7 - 3, dx = s % 7 - 3;
    float acc = 0.0f;
    for (int p = threadIdx.x; p < NPIX; p += 256) {
        int y = p >> 6, x = p & 63;
        int ys = y + dy; ys = ys < 0 ? 0 : (ys > 63 ? 63 : ys);
        int xs = x + dx; xs = xs < 0 ? 0 : (xs > 63 ? 63 : xs);
        const _Float16* a = rn + ((size_t)b * NPIX + p) * 32;
        const _Float16* c = vn + ((size_t)b * NPIX + ys * 64 + xs) * 32;
        float d = 0.0f;
#pragma unroll
        for (int k = 0; k < 32; ++k) d += (float)a[k] * (float)c[k];
        acc += w2[b * NPIX + p] * d;
    }
    sm[threadIdx.x] = acc; __syncthreads();
    for (int off = 128; off > 0; off >>= 1) {
        if (threadIdx.x < off) sm[threadIdx.x] += sm[threadIdx.x + off];
        __syncthreads();
    }
    if (threadIdx.x == 0) logits[bs] = sm[0] * 0.17677669529663689f;   // 1/sqrt(32)
}

__global__ void softmax_kernel(const float* __restrict__ logits, const float* __restrict__ log_temp,
                               float* __restrict__ dout, float* __restrict__ coarse) {
    int b = threadIdx.x;
    if (b >= 128) return;
    float T = fmaxf(expf(log_temp[0]), 1e-3f);
    float l[49];
    float m = -1e30f;
    for (int s = 0; s < 49; ++s) {
        l[s] = logits[b * 49 + s];
        dout[1024 + b * 49 + s] = l[s];        // raw (pre-temperature) logits output
        float t = l[s] / T;
        if (t > m) m = t;
    }
    float sum = 0.0f;
    for (int s = 0; s < 49; ++s) sum += expf(l[s] / T - m);
    float cdx = 0.0f, cdy = 0.0f, conf = 0.0f;
    for (int s = 0; s < 49; ++s) {
        float p = expf(l[s] / T - m) / sum;
        cdx += p * (float)(s % 7 - 3);
        cdy += p * (float)(s / 7 - 3);
        conf = fmaxf(conf, p);
    }
    coarse[b * 2] = cdx; coarse[b * 2 + 1] = cdy;
    dout[256 + b] = cdx;
    dout[384 + b] = cdy;
    dout[896 + b] = conf;
}

// ---------------------------------------------------------------------------
// Gaussian-weighted pooling: pooled[b, dstoff + c] = sum_p feat[b,p,c]*gaussN(p)
// ---------------------------------------------------------------------------
__global__ void pool_kernel(const _Float16* __restrict__ feat, const float* __restrict__ gsum,
                            float* __restrict__ pooled, int dstoff) {
    __shared__ float sm[256];
    int b = blockIdx.x;
    int c = threadIdx.x & 31, gi = threadIdx.x >> 5;
    float gs = gsum[0];
    float acc = 0.0f;
    for (int p = gi; p < NPIX; p += 8)
        acc += (float)feat[((size_t)b * NPIX + p) * 32 + c] * (gauss_raw(p) / gs);
    sm[threadIdx.x] = acc; __syncthreads();
    for (int off = 4; off > 0; off >>= 1) {
        if (gi < off) sm[threadIdx.x] += sm[threadIdx.x + off * 32];
        __syncthreads();
    }
    if (gi == 0) pooled[b * 114 + dstoff + c] = sm[c];
}

__global__ void assemble_kernel(float* __restrict__ pooled, const float* __restrict__ coarse,
                                const int* __restrict__ band_idx, const float* __restrict__ band_emb) {
    int b = blockIdx.x, t = threadIdx.x;
    float* pb = pooled + b * 114;
    if (t < 32)       pb[64 + t] = pb[t] - pb[32 + t];
    else if (t == 32) pb[96] = coarse[b * 2];
    else if (t == 33) pb[97] = coarse[b * 2 + 1];
    else if (t >= 34 && t < 50) pb[98 + t - 34] = band_emb[band_idx[b] * 16 + (t - 34)];
}

__global__ void mlp_kernel(const float* __restrict__ in, const float* __restrict__ w,
                           const float* __restrict__ bias, float* __restrict__ outv,
                           int K, int N, int doGelu) {
    int b = blockIdx.x, j = threadIdx.x;
    if (j >= N) return;
    float s = bias[j];
    const float* ib = in + b * K;
    const float* wj = w + j * K;
    for (int k = 0; k < K; ++k) s += ib[k] * wj[k];
    if (doGelu) s = gelu_erf(s);
    outv[b * N + j] = s;
}

__global__ void final_kernel(const float* __restrict__ residual, const float* __restrict__ coarse,
                             const float* __restrict__ p2s, float* __restrict__ dout) {
    int b = threadIdx.x;
    if (b >= 128) return;
    float cdx = coarse[b * 2], cdy = coarse[b * 2 + 1];
    float dx = cdx + residual[b * 3];
    float dy = cdy + residual[b * 3 + 1];
    float ls = fminf(fmaxf(residual[b * 3 + 2], -6.0f), 3.0f);
    const float* m = p2s + b * 4;
    dout[b]       = dx;
    dout[128 + b] = dy;
    dout[512 + b * 2]     = m[0] * dx + m[1] * dy;
    dout[512 + b * 2 + 1] = m[2] * dx + m[3] * dy;
    dout[768 + b] = ls;
}

// ---------------------------------------------------------------------------
extern "C" void kernel_launch(void* const* d_in, const int* in_sizes, int n_in,
                              void* d_out, int out_size, void* d_ws, size_t ws_size,
                              hipStream_t stream) {
    const float* rubin_patch = (const float*)d_in[0];
    const float* vis_patch   = (const float*)d_in[1];
    const float* pixel_to_sky= (const float*)d_in[2];
    const int*   band_idx    = (const int*)d_in[3];
    const float* rubin_conv0 = (const float*)d_in[4];
    const float* rubin_convs = (const float*)d_in[5];
    const float* rubin_gn_g  = (const float*)d_in[6];
    const float* rubin_gn_b  = (const float*)d_in[7];
    const float* vis_conv0   = (const float*)d_in[8];
    const float* vis_convs   = (const float*)d_in[9];
    const float* vis_gn_g    = (const float*)d_in[10];
    const float* vis_gn_b    = (const float*)d_in[11];
    const float* rubin_proj  = (const float*)d_in[12];
    const float* vis_proj    = (const float*)d_in[13];
    const float* log_temp    = (const float*)d_in[14];
    const float* band_emb    = (const float*)d_in[15];
    const float* w1 = (const float*)d_in[16];
    const float* b1 = (const float*)d_in[17];
    const float* w2 = (const float*)d_in[18];
    const float* b2 = (const float*)d_in[19];
    const float* w3 = (const float*)d_in[20];
    const float* b3 = (const float*)d_in[21];
    float* dout = (float*)d_out;

    char* ws = (char*)d_ws;
    size_t off = 0;
    auto take = [&](size_t bytes) {
        size_t r = off;
        off += (bytes + 255) & ~(size_t)255;
        return r;
    };
    _Float16* actA      = (_Float16*)(ws + take((size_t)NELEM * 2));
    _Float16* actB      = (_Float16*)(ws + take((size_t)NELEM * 2));
    float*    convOut   = (float*)   (ws + take((size_t)NELEM * 4));
    _Float16* rubinFeat = (_Float16*)(ws + take((size_t)NELEM * 2));
    _Float16* visFeat   = (_Float16*)(ws + take((size_t)NELEM * 2));
    _Float16* packedW   = (_Float16*)(ws + take(57344 * 2));
    float*    gnStats   = (float*)   (ws + take(512 * 2 * 4));
    float*    energy    = (float*)   (ws + take((size_t)NBPIX * 4));
    float*    w2buf     = (float*)   (ws + take((size_t)NBPIX * 4));
    float*    gsumP     = (float*)   (ws + take(256));
    float*    swsumP    = (float*)   (ws + take(128 * 4));
    float*    logitsB   = (float*)   (ws + take(6272 * 4));
    float*    coarseB   = (float*)   (ws + take(128 * 2 * 4));
    float*    pooled    = (float*)   (ws + take(128 * 114 * 4));
    float*    h1        = (float*)   (ws + take(128 * 128 * 4));
    float*    h2        = (float*)   (ws + take(128 * 128 * 4));
    float*    resid     = (float*)   (ws + take(128 * 3 * 4));
    _Float16* rubinN    = actA;   // reuse after encoders finish
    _Float16* visN      = actB;

    // ---- pack all WMMA weights (6 conv layers + 2 projections)
    pack_kernel<<<(57344 + 255) / 256, 256, 0, stream>>>(rubin_convs, vis_convs,
                                                         rubin_proj, vis_proj, packedW);

    // ---- rubin encoder
    conv0_kernel<<<65536, 256, 0, stream>>>(rubin_patch, rubin_conv0, 6, convOut);
    gn_stats_kernel<<<512, 256, 0, stream>>>(convOut, gnStats);
    gn_apply_kernel<<<65536, 256, 0, stream>>>(convOut, gnStats, rubin_gn_g, rubin_gn_b, actA);
    {
        _Float16* cur = actA;
        for (int l = 0; l < 3; ++l) {
            conv_wmma_kernel<9><<<4096, 256, 0, stream>>>(cur, packedW + l * 9216, convOut);
            gn_stats_kernel<<<512, 256, 0, stream>>>(convOut, gnStats);
            _Float16* dst = (l == 2) ? rubinFeat : ((cur == actA) ? actB : actA);
            gn_apply_kernel<<<65536, 256, 0, stream>>>(convOut, gnStats,
                rubin_gn_g + (l + 1) * 32, rubin_gn_b + (l + 1) * 32, dst);
            cur = dst;
        }
    }

    // ---- vis encoder
    conv0_kernel<<<65536, 256, 0, stream>>>(vis_patch, vis_conv0, 1, convOut);
    gn_stats_kernel<<<512, 256, 0, stream>>>(convOut, gnStats);
    gn_apply_kernel<<<65536, 256, 0, stream>>>(convOut, gnStats, vis_gn_g, vis_gn_b, actA);
    {
        _Float16* cur = actA;
        for (int l = 0; l < 3; ++l) {
            conv_wmma_kernel<9><<<4096, 256, 0, stream>>>(cur, packedW + (3 + l) * 9216, convOut);
            gn_stats_kernel<<<512, 256, 0, stream>>>(convOut, gnStats);
            _Float16* dst = (l == 2) ? visFeat : ((cur == actA) ? actB : actA);
            gn_apply_kernel<<<65536, 256, 0, stream>>>(convOut, gnStats,
                vis_gn_g + (l + 1) * 32, vis_gn_b + (l + 1) * 32, dst);
            cur = dst;
        }
    }

    // ---- projections + l2norm (WMMA, 1 tap)
    conv_wmma_kernel<1><<<4096, 256, 0, stream>>>(rubinFeat, packedW + 55296, convOut);
    l2norm_kernel<<<2048, 256, 0, stream>>>(convOut, rubinN, energy);
    conv_wmma_kernel<1><<<4096, 256, 0, stream>>>(visFeat, packedW + 55296 + 1024, convOut);
    l2norm_kernel<<<2048, 256, 0, stream>>>(convOut, visN, nullptr);

    // ---- spatial weights
    gsum_kernel<<<1, 256, 0, stream>>>(gsumP);
    swsum_kernel<<<128, 256, 0, stream>>>(energy, gsumP, swsumP);
    sw_kernel<<<2048, 256, 0, stream>>>(energy, gsumP, swsumP, w2buf);

    // ---- cost volume + softmax / soft-argmax
    cost_kernel<<<6272, 256, 0, stream>>>(rubinN, visN, w2buf, logitsB);
    softmax_kernel<<<1, 128, 0, stream>>>(logitsB, log_temp, dout, coarseB);

    // ---- pooling + MLP head
    pool_kernel<<<128, 256, 0, stream>>>(rubinFeat, gsumP, pooled, 0);
    pool_kernel<<<128, 256, 0, stream>>>(visFeat, gsumP, pooled, 32);
    assemble_kernel<<<128, 64, 0, stream>>>(pooled, coarseB, band_idx, band_emb);
    mlp_kernel<<<128, 128, 0, stream>>>(pooled, w1, b1, h1, 114, 128, 1);
    mlp_kernel<<<128, 128, 0, stream>>>(h1, w2, b2, h2, 128, 128, 1);
    mlp_kernel<<<128, 32, 0, stream>>>(h2, w3, b3, resid, 128, 3, 0);
    final_kernel<<<1, 128, 0, stream>>>(resid, coarseB, pixel_to_sky, dout);
}
